// LRULayer_9732395892794
// MI455X (gfx1250) — matrix operand order repaired and checked by reading, not compiled, not run on hardware
//
#include <hip/hip_runtime.h>
#include <hip/hip_bf16.h>
#include <math.h>

typedef __bf16 bf16;
typedef bf16  v8bf  __attribute__((ext_vector_type(8)));
typedef bf16  v16bf __attribute__((ext_vector_type(16)));
typedef float v8f   __attribute__((ext_vector_type(8)));

constexpr int BATCH   = 8;
constexpr int LSEQ    = 2048;
constexpr int DMODEL  = 512;
constexpr int HID     = 1024;
constexpr int MTOK    = BATCH * LSEQ;   // 16384
constexpr int N1      = 2 * HID;        // 2048 (h_r | h_i stacked)
constexpr int K1      = DMODEL;         // 512
constexpr int N2      = DMODEL;         // 512
constexpr int K2      = 2 * HID;        // 2048

// workspace layout (bytes)
constexpr size_t OFF_H32   = 0;                                        // f32 h   [MTOK, 2048]
constexpr size_t OFF_H16   = OFF_H32 + (size_t)MTOK * N1 * 4;          // bf16 h  [MTOK, 2048]
constexpr size_t OFF_XB    = OFF_H16 + (size_t)MTOK * N1 * 2;          // bf16 x  [MTOK, 512]
constexpr size_t OFF_W1    = OFF_XB  + (size_t)MTOK * K1 * 2;          // bf16 Win [2048, 512]
constexpr size_t OFF_W2    = OFF_W1  + (size_t)N1 * K1 * 2;            // bf16 Wout' [512, 2048]
constexpr size_t OFF_SCALE = OFF_W2  + (size_t)N2 * K2 * 2;            // f32 [2048]
constexpr size_t OFF_BIASG = OFF_SCALE + (size_t)N1 * 4;               // f32 [2048]

// ---------------- fragment loads (per CDNA5 ISA §7.12.2 16-bit layouts) -----
// A 16x32: lanes 0-15 / 16-31 both hold rows M=0..15; low half holds K{0..7,16..23},
// high half K{8..15,24..31}.  Two 16B contiguous chunks per lane.
__device__ __forceinline__ v16bf load_a(const bf16* __restrict__ A, int lda,
                                        int row, int k0, int hi) {
  const bf16* p = A + (size_t)row * lda + k0 + hi * 8;
  v8bf lo = *(const v8bf*)(p);
  v8bf hh = *(const v8bf*)(p + 16);
  return __builtin_shufflevector(lo, hh, 0,1,2,3,4,5,6,7,8,9,10,11,12,13,14,15);
}
// B 32x16 as W rows (C = A * W^T): lane n (0-15) holds col n, K{0..7,16..23};
// lanes 16-31 hold K{8..15,24..31}.  Two 16B contiguous chunks per lane.
__device__ __forceinline__ v16bf load_b(const bf16* __restrict__ W, int ldw,
                                        int wrow, int k0, int hi) {
  const bf16* p = W + (size_t)wrow * ldw + k0 + hi * 8;
  v8bf lo = *(const v8bf*)(p);
  v8bf hh = *(const v8bf*)(p + 16);
  return __builtin_shufflevector(lo, hh, 0,1,2,3,4,5,6,7,8,9,10,11,12,13,14,15);
}

// ---------------- WMMA GEMM: C[M, N] = A[M,K](bf16) * W[N,K](bf16)^T --------
// block tile 128x256, 8 waves (wave32) in 2(M) x 4(N); wave tile 64x64
// (4 M-frags x 4 N-frags -> 16 v_wmma per K-step, 16 b128 loads: 1:1).
// EPI 0: out = acc*scale[col] + bias[col]            (input projection, *gamma)
// EPI 1: out = acc + bias[col] + addin[row*N + col]  (output proj + residual)
template <int KDIM, int EPI>
__global__ __launch_bounds__(256) void
gemm_wmma(const bf16* __restrict__ A, const bf16* __restrict__ W,
          float* __restrict__ C, int ncols,
          const float* __restrict__ scale, const float* __restrict__ bias,
          const float* __restrict__ addin) {
  const int lane = threadIdx.x & 31;
  const int wave = threadIdx.x >> 5;
  const int wm   = wave & 1;        // 2 M-waves
  const int wn   = wave >> 1;       // 4 N-waves
  const int l15  = lane & 15;
  const int hi   = lane >> 4;

  const int m_wave = blockIdx.y * 128 + wm * 64;
  const int n_wave = blockIdx.x * 256 + wn * 64;

  v8f acc[4][4];
  v8f zero = {0.f, 0.f, 0.f, 0.f, 0.f, 0.f, 0.f, 0.f};
#pragma unroll
  for (int i = 0; i < 4; ++i)
#pragma unroll
    for (int j = 0; j < 4; ++j) acc[i][j] = zero;

  for (int k0 = 0; k0 < KDIM; k0 += 32) {
    // prefetch the streams two K-steps ahead (uniform guard; lanes cover rows)
    if (k0 + 64 < KDIM) {
      __builtin_prefetch(A + (size_t)(m_wave + l15 + hi * 16) * KDIM + k0 + 64, 0, 1);
      __builtin_prefetch(W + (size_t)(n_wave + l15 + hi * 16) * KDIM + k0 + 64, 0, 1);
    }
    v16bf a[4], b[4];
#pragma unroll
    for (int i = 0; i < 4; ++i)
      a[i] = load_a(A, KDIM, m_wave + i * 16 + l15, k0, hi);
#pragma unroll
    for (int j = 0; j < 4; ++j)
      b[j] = load_b(W, KDIM, n_wave + j * 16 + l15, k0, hi);
#pragma unroll
    for (int i = 0; i < 4; ++i)
#pragma unroll
      for (int j = 0; j < 4; ++j)
        acc[i][j] = __builtin_amdgcn_wmma_f32_16x16x32_bf16(
            false, a[i], false, b[j], (short)0, acc[i][j], false, false);
  }

  // C frag: VGPR r -> row r (lanes 0-15) / r+8 (lanes 16-31), col = lane&15
#pragma unroll
  for (int i = 0; i < 4; ++i)
#pragma unroll
    for (int j = 0; j < 4; ++j)
#pragma unroll
      for (int r = 0; r < 8; ++r) {
        int row = m_wave + i * 16 + r + hi * 8;
        int col = n_wave + j * 16 + l15;
        float v = acc[i][j][r];
        if constexpr (EPI == 0) {
          v = v * scale[col] + bias[col];
        } else {
          v = v + bias[col] + addin[(size_t)row * ncols + col];
        }
        C[(size_t)row * ncols + col] = v;
      }
}

// ---------------- sequential complex scan, channel-parallel ------------------
__global__ __launch_bounds__(256) void
scan_kernel(const float* __restrict__ H32, bf16* __restrict__ H16,
            const float* __restrict__ mask,
            const float* __restrict__ nu_log, const float* __restrict__ theta_log) {
  int g = blockIdx.x * blockDim.x + threadIdx.x;  // 0..8191
  int b = g >> 10;
  int d = g & (HID - 1);
  float nu  = __expf(nu_log[d]);
  float th  = __expf(theta_log[d]);
  float mag = __expf(-nu);
  float lr  = mag * __cosf(th);
  float li  = mag * __sinf(th);
  float cr = 0.f, ci = 0.f, mprev = 1.f;
  size_t rowbase = (size_t)b * LSEQ;
#pragma unroll 4
  for (int l = 0; l < LSEQ; ++l) {
    size_t off = (rowbase + l) * (size_t)N1;
    float hr = H32[off + d];
    float hv = H32[off + HID + d];
    float nr = hr + mprev * (lr * cr - li * ci);
    float ni = hv + mprev * (lr * ci + li * cr);
    cr = nr; ci = ni;
    H16[off + d]        = (bf16)cr;
    H16[off + HID + d]  = (bf16)ci;
    mprev = mask[rowbase + l];
  }
}

// ---------------- LayerNorm, one wave32 per row of 512 -----------------------
__global__ __launch_bounds__(256) void
ln_kernel(float* __restrict__ Y, const float* __restrict__ ln_w,
          const float* __restrict__ ln_b) {
  int lane = threadIdx.x & 31;
  int row  = blockIdx.x * 8 + (threadIdx.x >> 5);
  size_t base = (size_t)row * DMODEL;
  float vals[16];
  float s = 0.f, s2 = 0.f;
#pragma unroll
  for (int i = 0; i < 16; ++i) {
    float v = Y[base + lane + i * 32];
    vals[i] = v; s += v; s2 += v * v;
  }
#pragma unroll
  for (int off = 16; off >= 1; off >>= 1) {
    s  += __shfl_xor(s,  off, 32);
    s2 += __shfl_xor(s2, off, 32);
  }
  float mu   = s * (1.f / DMODEL);
  float var  = s2 * (1.f / DMODEL) - mu * mu;
  float rstd = rsqrtf(var + 1e-5f);
#pragma unroll
  for (int i = 0; i < 16; ++i) {
    int c = lane + i * 32;
    Y[base + c] = (vals[i] - mu) * rstd * ln_w[c] + ln_b[c];
  }
}

// ---------------- small converts / parameter prep ----------------------------
__global__ void cvt_x_kernel(const float* __restrict__ x, bf16* __restrict__ xb, int n) {
  int i = blockIdx.x * blockDim.x + threadIdx.x;
  if (i < n) xb[i] = (bf16)x[i];
}
__global__ void cvt_w1_kernel(const float* __restrict__ Wr, const float* __restrict__ Wi,
                              bf16* __restrict__ Wb) {
  int i = blockIdx.x * blockDim.x + threadIdx.x;  // N1*K1
  if (i >= N1 * K1) return;
  int n = i / K1, k = i - n * K1;
  float v = (n < HID) ? Wr[n * K1 + k] : Wi[(n - HID) * K1 + k];
  Wb[i] = (bf16)v;
}
__global__ void cvt_w2_kernel(const float* __restrict__ Wr, const float* __restrict__ Wi,
                              bf16* __restrict__ Wb) {
  int i = blockIdx.x * blockDim.x + threadIdx.x;  // N2*K2
  if (i >= N2 * K2) return;
  int e = i / K2, j = i - e * K2;
  float v = (j < HID) ? Wr[e * HID + j] : -Wi[e * HID + (j - HID)];
  Wb[i] = (bf16)v;
}
__global__ void prep_params_kernel(const float* __restrict__ gamma_log,
                                   const float* __restrict__ b_in_r,
                                   const float* __restrict__ b_in_i,
                                   float* __restrict__ scale, float* __restrict__ biasg) {
  int n = blockIdx.x * blockDim.x + threadIdx.x;
  if (n >= N1) return;
  int d = n & (HID - 1);
  float g = __expf(gamma_log[d]);
  scale[n] = g;
  biasg[n] = (n < HID ? b_in_r[d] : b_in_i[d]) * g;
}

extern "C" void kernel_launch(void* const* d_in, const int* in_sizes, int n_in,
                              void* d_out, int out_size, void* d_ws, size_t ws_size,
                              hipStream_t stream) {
  const float* x         = (const float*)d_in[0];
  const float* mask      = (const float*)d_in[1];
  const float* nu_log    = (const float*)d_in[2];
  const float* theta_log = (const float*)d_in[3];
  const float* gamma_log = (const float*)d_in[4];
  const float* W_in_r    = (const float*)d_in[5];
  const float* W_in_i    = (const float*)d_in[6];
  const float* b_in_r    = (const float*)d_in[7];
  const float* b_in_i    = (const float*)d_in[8];
  const float* W_out_r   = (const float*)d_in[9];
  const float* W_out_i   = (const float*)d_in[10];
  const float* b_out_r   = (const float*)d_in[11];
  const float* ln_w      = (const float*)d_in[13];
  const float* ln_b      = (const float*)d_in[14];

  char*  ws    = (char*)d_ws;
  float* H32   = (float*)(ws + OFF_H32);
  bf16*  H16   = (bf16*) (ws + OFF_H16);
  bf16*  XB    = (bf16*) (ws + OFF_XB);
  bf16*  W1B   = (bf16*) (ws + OFF_W1);
  bf16*  W2B   = (bf16*) (ws + OFF_W2);
  float* SCALE = (float*)(ws + OFF_SCALE);
  float* BIASG = (float*)(ws + OFF_BIASG);
  float* OUT   = (float*)d_out;

  // converts + parameter prep
  cvt_x_kernel<<<(MTOK * K1 + 255) / 256, 256, 0, stream>>>(x, XB, MTOK * K1);
  cvt_w1_kernel<<<(N1 * K1 + 255) / 256, 256, 0, stream>>>(W_in_r, W_in_i, W1B);
  cvt_w2_kernel<<<(N2 * K2 + 255) / 256, 256, 0, stream>>>(W_out_r, W_out_i, W2B);
  prep_params_kernel<<<(N1 + 255) / 256, 256, 0, stream>>>(gamma_log, b_in_r, b_in_i,
                                                           SCALE, BIASG);
  // h = (x @ W_in^T + b_in) * gamma   (f32 out for the scan)
  gemm_wmma<K1, 0><<<dim3(N1 / 256, MTOK / 128), 256, 0, stream>>>(
      XB, W1B, H32, N1, SCALE, BIASG, nullptr);
  // complex linear recurrence along L; emits bf16 h for the output GEMM
  scan_kernel<<<(BATCH * HID) / 256, 256, 0, stream>>>(H32, H16, mask, nu_log, theta_log);
  // y = Re(h @ W_out^T) + b_out_r + x
  gemm_wmma<K2, 1><<<dim3(N2 / 256, MTOK / 128), 256, 0, stream>>>(
      H16, W2B, OUT, N2, nullptr, b_out_r, x);
  // in-place LayerNorm
  ln_kernel<<<MTOK / 8, 256, 0, stream>>>(OUT, ln_w, ln_b);
}
